// ScoreGraphReconstructor_77824807404028
// MI455X (gfx1250) — matrix operand order repaired and checked by reading, not compiled, not run on hardware
//
#include <hip/hip_runtime.h>

// ---------------- problem constants ----------------
#define NN   50000      // nodes
#define EE   400000     // candidate edges
#define EA   450000     // edges + self loops
#define HID  256        // hidden dim
#define NHEAD 4
#define CDIM 64

typedef __attribute__((ext_vector_type(16))) _Float16 v16h;
typedef __attribute__((ext_vector_type(8)))  float    v8f;
typedef __attribute__((ext_vector_type(8)))  _Float16 f16x8;
typedef __attribute__((ext_vector_type(4)))  float    f32x4;

// ---------------- WMMA fragment loaders (CDNA5 16x16x32 f16 layouts) ----------------
// A (16x32, row-major, row stride ld halves): lanes 0-15 row=lane K={0..7,16..23},
// lanes 16-31 same row K={8..15,24..31}
__device__ inline v16h load_a_frag(const _Float16* __restrict__ base, int ld, int lane) {
    int r = lane & 15, hi = lane >> 4;
    f16x8 lo = *(const f16x8*)(base + (size_t)r * ld + hi * 8);
    f16x8 hv = *(const f16x8*)(base + (size_t)r * ld + 16 + hi * 8);
    v16h a;
#pragma unroll
    for (int j = 0; j < 8; ++j) { a[j] = lo[j]; a[8 + j] = hv[j]; }
    return a;
}

// B (32x16) sourced from transposed weight Bt[N][K]: element j holds K = hi*8+j (j<8), 16+hi*8+(j-8) (j>=8), col = lane&15
__device__ inline v16h load_b_frag(const _Float16* __restrict__ bt, int ldb, int lane) {
    int n = lane & 15, hi = lane >> 4;
    f16x8 lo = *(const f16x8*)(bt + (size_t)n * ldb + hi * 8);
    f16x8 hv = *(const f16x8*)(bt + (size_t)n * ldb + 16 + hi * 8);
    v16h b;
#pragma unroll
    for (int j = 0; j < 8; ++j) { b[j] = lo[j]; b[8 + j] = hv[j]; }
    return b;
}

__device__ inline v8f wmma_f16(v16h a, v16h b, v8f c) {
    return __builtin_amdgcn_wmma_f32_16x16x32_f16(false, a, false, b, (short)0, c, false, false);
}

// order-preserving float<->uint mapping for atomicMax-based segment max
__device__ inline unsigned enc_f(float f) {
    unsigned u = __float_as_uint(f);
    return (u & 0x80000000u) ? ~u : (u | 0x80000000u);
}
__device__ inline float dec_f(unsigned v) {
    return __uint_as_float((v & 0x80000000u) ? (v ^ 0x80000000u) : ~v);
}
#define ENC_NEG_INF 0x007FFFFFu   // enc(-inf)

__device__ inline float elu_f(float v) { return v > 0.f ? v : (__expf(v) - 1.f); }

// ---------------- small utility kernels ----------------
__global__ void k_f32_to_f16(const float* __restrict__ src, _Float16* __restrict__ dst, int n) {
    int i = blockIdx.x * 256 + threadIdx.x;
    if (i < n) dst[i] = (_Float16)src[i];
}

// Wt[n*K+k] = (f16)W[k*N+n]
__global__ void k_transpose_w(const float* __restrict__ W, _Float16* __restrict__ Wt, int K, int N) {
    int i = blockIdx.x * 256 + threadIdx.x;
    if (i >= K * N) return;
    int k = i / N, n = i - k * N;
    Wt[(size_t)n * K + k] = (_Float16)W[i];
}

__global__ void k_init_seg(float* __restrict__ acc, unsigned* __restrict__ emax,
                           float* __restrict__ denom, int nfeat, int nh) {
    int i = blockIdx.x * 256 + threadIdx.x;
    if (i < nfeat) acc[i] = 0.f;
    if (i < nh) { emax[i] = ENC_NEG_INF; denom[i] = 0.f; }
}

// ---------------- node transform GEMM: Y[M,256] = A(f16)[M,K] @ Bt(f16)[256,K]^T + bias ----------------
// one wave -> one 16x64 output tile; grid.x covers M tiles (8 per block), grid.y = 256/64
__global__ void k_gemm_wmma(const _Float16* __restrict__ A, const _Float16* __restrict__ Bt,
                            const float* __restrict__ bias, float* __restrict__ Y,
                            int Mtiles, int K) {
    int wave = threadIdx.x >> 5, lane = threadIdx.x & 31;
    int mt = blockIdx.x * 8 + wave;
    if (mt >= Mtiles) return;
    int n0 = blockIdx.y * 64;
    v8f c0 = {}, c1 = {}, c2 = {}, c3 = {};
    const _Float16* arow = A + (size_t)mt * 16 * K;
    for (int k0 = 0; k0 < K; k0 += 32) {
        v16h a  = load_a_frag(arow + k0, K, lane);
        v16h b0 = load_b_frag(Bt + (size_t)(n0 +  0) * K + k0, K, lane);
        v16h b1 = load_b_frag(Bt + (size_t)(n0 + 16) * K + k0, K, lane);
        v16h b2 = load_b_frag(Bt + (size_t)(n0 + 32) * K + k0, K, lane);
        v16h b3 = load_b_frag(Bt + (size_t)(n0 + 48) * K + k0, K, lane);
        c0 = wmma_f16(a, b0, c0);
        c1 = wmma_f16(a, b1, c1);
        c2 = wmma_f16(a, b2, c2);
        c3 = wmma_f16(a, b3, c3);
    }
    int n = lane & 15, hi = lane >> 4;
#pragma unroll
    for (int i = 0; i < 8; ++i) {
        size_t row = (size_t)mt * 16 + hi * 8 + i;
        int col = n0 + n;
        Y[row * HID + col +  0] = c0[i] + bias[col +  0];
        Y[row * HID + col + 16] = c1[i] + bias[col + 16];
        Y[row * HID + col + 32] = c2[i] + bias[col + 32];
        Y[row * HID + col + 48] = c3[i] + bias[col + 48];
    }
}

// ---------------- attention pass 1: e[edge][h], segment max ----------------
__global__ void k_attn_score(const float* __restrict__ xl, const float* __restrict__ xr,
                             const float* __restrict__ att, const int* __restrict__ ei,
                             float* __restrict__ ebuf, unsigned* __restrict__ emax) {
    int wave = threadIdx.x >> 5, lane = threadIdx.x & 31;
    int e = blockIdx.x * 8 + wave;
    if (e >= EA) return;
    int s, d;
    if (e < EE) { s = ei[e]; d = ei[EE + e]; } else { s = d = e - EE; }
    int f0 = lane * 8;
    const f32x4* pl = (const f32x4*)(xl + (size_t)s * HID + f0);
    const f32x4* pr = (const f32x4*)(xr + (size_t)d * HID + f0);
    f32x4 l0 = pl[0], l1 = pl[1], r0 = pr[0], r1 = pr[1];
    float part = 0.f;
#pragma unroll
    for (int j = 0; j < 4; ++j) {
        float m0 = l0[j] + r0[j]; m0 = m0 > 0.f ? m0 : 0.2f * m0;
        float m1 = l1[j] + r1[j]; m1 = m1 > 0.f ? m1 : 0.2f * m1;
        part += m0 * att[f0 + j] + m1 * att[f0 + 4 + j];
    }
    part += __shfl_xor(part, 1, 32);
    part += __shfl_xor(part, 2, 32);
    part += __shfl_xor(part, 4, 32);
    if ((lane & 7) == 0) {
        int h = lane >> 3;
        ebuf[(size_t)e * NHEAD + h] = part;
        atomicMax(&emax[(size_t)d * NHEAD + h], enc_f(part));
    }
}

// ---------------- attention pass 2: a=exp(e-max); denom += a; acc += a*xl[src] ----------------
__global__ void k_attn_scatter(const float* __restrict__ xl, const float* __restrict__ ebuf,
                               const unsigned* __restrict__ emax, const int* __restrict__ ei,
                               float* __restrict__ denom, float* __restrict__ acc) {
    int wave = threadIdx.x >> 5, lane = threadIdx.x & 31;
    int e = blockIdx.x * 8 + wave;
    if (e >= EA) return;
    int s, d;
    if (e < EE) { s = ei[e]; d = ei[EE + e]; } else { s = d = e - EE; }
    int h = lane >> 3;
    float ev = ebuf[(size_t)e * NHEAD + h];
    float mx = dec_f(emax[(size_t)d * NHEAD + h]);
    float a = __expf(ev - mx);
    if ((lane & 7) == 0) atomicAdd(&denom[(size_t)d * NHEAD + h], a);
    int f0 = lane * 8;
    const float* ps = xl + (size_t)s * HID + f0;
    float* pd = acc + (size_t)d * HID + f0;
#pragma unroll
    for (int j = 0; j < 8; ++j) atomicAdd(&pd[j], a * ps[j]);
}

// ---------------- per-node finish: h = acc/denom + bo [+ elu] ----------------
__global__ void k_node_finish(const float* __restrict__ acc, const float* __restrict__ denom,
                              const float* __restrict__ bo, float* __restrict__ hout, int applyElu) {
    int i = blockIdx.x * 256 + threadIdx.x;            // grid == NN*HID exactly
    int f = i & (HID - 1);
    int nidx = i >> 8;
    int h = f >> 6;
    float v = acc[i] / denom[(size_t)nidx * NHEAD + h] + bo[f];
    if (applyElu) v = elu_f(v);
    hout[i] = v;
}

// ---------------- fused edge classifier MLP: 512 -> 256 -> 128 -> 5 ----------------
// block = 256 threads / 8 waves handles a 16-edge tile; A + intermediates live in LDS
__global__ void k_edge_mlp(const _Float16* __restrict__ h3, const int* __restrict__ ei,
                           const _Float16* __restrict__ W1t, const float* __restrict__ b1,
                           const _Float16* __restrict__ W2t, const float* __restrict__ b2,
                           const float* __restrict__ W3, const float* __restrict__ b3,
                           float* __restrict__ out) {
    __shared__ __align__(16) _Float16 sA[16 * 520];    // ef tile, padded stride
    __shared__ __align__(16) _Float16 sZ1[16 * 264];
    __shared__ __align__(16) _Float16 sZ2[16 * 136];
    int tid = threadIdx.x, wave = tid >> 5, lane = tid & 31;
    int e0 = blockIdx.x * 16;

    // gather ef = concat(h3[row], h3[col]) : 16 threads per edge, 32 halves each
    {
        int m = tid >> 4, t = tid & 15;
        int eg = e0 + m;
        int rs = ei[eg], cs = ei[EE + eg];
#pragma unroll
        for (int v = 0; v < 4; ++v) {
            int off = t * 32 + v * 8;
            const f16x8* src = (off < 256)
                ? (const f16x8*)(h3 + (size_t)rs * HID + off)
                : (const f16x8*)(h3 + (size_t)cs * HID + (off - 256));
            *(f16x8*)(sA + m * 520 + off) = *src;
        }
    }
    __syncthreads();

    int n = lane & 15, hi = lane >> 4;

    // layer 1: [16x512] @ [512x256], 16 n-tiles -> 2 per wave
#pragma unroll
    for (int nt = 0; nt < 2; ++nt) {
        int n0 = (wave * 2 + nt) * 16;
        v8f c = {};
        for (int k0 = 0; k0 < 512; k0 += 32) {
            v16h a = load_a_frag(sA + k0, 520, lane);
            v16h b = load_b_frag(W1t + (size_t)n0 * 512 + k0, 512, lane);
            c = wmma_f16(a, b, c);
        }
#pragma unroll
        for (int i = 0; i < 8; ++i) {
            int row = hi * 8 + i, col = n0 + n;
            sZ1[row * 264 + col] = (_Float16)elu_f(c[i] + b1[col]);
        }
    }
    __syncthreads();

    // layer 2: [16x256] @ [256x128], 8 n-tiles -> 1 per wave
    {
        int n0 = wave * 16;
        v8f c = {};
        for (int k0 = 0; k0 < 256; k0 += 32) {
            v16h a = load_a_frag(sZ1 + k0, 264, lane);
            v16h b = load_b_frag(W2t + (size_t)n0 * 256 + k0, 256, lane);
            c = wmma_f16(a, b, c);
        }
#pragma unroll
        for (int i = 0; i < 8; ++i) {
            int row = hi * 8 + i, col = n0 + n;
            sZ2[row * 136 + col] = (_Float16)elu_f(c[i] + b2[col]);
        }
    }
    __syncthreads();

    // layer 3: [16x128] @ [128x5] scalar (tiny)
    if (tid < 80) {
        int m = tid / 5, cls = tid - m * 5;
        float s = b3[cls];
        for (int k = 0; k < 128; ++k) s += (float)sZ2[m * 136 + k] * W3[k * 5 + cls];
        out[(size_t)(e0 + m) * 5 + cls] = s;
    }
}

// ---------------- host side ----------------
extern "C" void kernel_launch(void* const* d_in, const int* in_sizes, int n_in,
                              void* d_out, int out_size, void* d_ws, size_t ws_size,
                              hipStream_t stream) {
    (void)in_sizes; (void)n_in; (void)out_size; (void)ws_size;
    const float* x  = (const float*)d_in[0];
    const int*   ei = (const int*)d_in[1];
    const float* Wl[3]  = {(const float*)d_in[2],  (const float*)d_in[8],  (const float*)d_in[14]};
    const float* bl[3]  = {(const float*)d_in[3],  (const float*)d_in[9],  (const float*)d_in[15]};
    const float* Wr[3]  = {(const float*)d_in[4],  (const float*)d_in[10], (const float*)d_in[16]};
    const float* br[3]  = {(const float*)d_in[5],  (const float*)d_in[11], (const float*)d_in[17]};
    const float* att[3] = {(const float*)d_in[6],  (const float*)d_in[12], (const float*)d_in[18]};
    const float* bo[3]  = {(const float*)d_in[7],  (const float*)d_in[13], (const float*)d_in[19]};
    const float* We1 = (const float*)d_in[20]; const float* be1 = (const float*)d_in[21];
    const float* We2 = (const float*)d_in[22]; const float* be2 = (const float*)d_in[23];
    const float* We3 = (const float*)d_in[24]; const float* be3 = (const float*)d_in[25];
    float* out = (float*)d_out;

    char* p = (char*)d_ws;
    auto carve = [&](size_t bytes) -> void* {
        void* r = (void*)p;
        p += (bytes + 255) & ~(size_t)255;
        return r;
    };
    float*     hcur  = (float*)carve((size_t)NN * HID * 4);
    float*     xl    = (float*)carve((size_t)NN * HID * 4);
    float*     xr    = (float*)carve((size_t)NN * HID * 4);
    float*     acc   = (float*)carve((size_t)NN * HID * 4);
    _Float16*  hh    = (_Float16*)carve((size_t)NN * HID * 2);
    float*     ebuf  = (float*)carve((size_t)EA * NHEAD * 4);
    unsigned*  emax  = (unsigned*)carve((size_t)NN * NHEAD * 4);
    float*     denom = (float*)carve((size_t)NN * NHEAD * 4);
    _Float16*  Wlt[3], *Wrt[3];
    for (int i = 0; i < 3; ++i) {
        Wlt[i] = (_Float16*)carve((size_t)HID * HID * 2);
        Wrt[i] = (_Float16*)carve((size_t)HID * HID * 2);
    }
    _Float16* W1t = (_Float16*)carve((size_t)512 * 256 * 2);
    _Float16* W2t = (_Float16*)carve((size_t)256 * 128 * 2);

    // pre-transpose + f16-convert all GEMM weights (L2-resident, tiny)
    for (int i = 0; i < 3; ++i) {
        k_transpose_w<<<(HID * HID + 255) / 256, 256, 0, stream>>>(Wl[i], Wlt[i], HID, HID);
        k_transpose_w<<<(HID * HID + 255) / 256, 256, 0, stream>>>(Wr[i], Wrt[i], HID, HID);
    }
    k_transpose_w<<<(512 * 256 + 255) / 256, 256, 0, stream>>>(We1, W1t, 512, 256);
    k_transpose_w<<<(256 * 128 + 255) / 256, 256, 0, stream>>>(We2, W2t, 256, 128);

    const int Mtiles  = NN / 16;                  // 3125 (exact)
    const dim3 gGemm((Mtiles + 7) / 8, 4);
    const int nElem   = NN * HID;                 // 12.8M, multiple of 256
    const int edgeBlk = EA / 8;                   // 56250 (exact)

    for (int li = 0; li < 3; ++li) {
        const float* hin = (li == 0) ? x : hcur;
        k_f32_to_f16<<<nElem / 256, 256, 0, stream>>>(hin, hh, nElem);
        k_gemm_wmma<<<gGemm, 256, 0, stream>>>(hh, Wlt[li], bl[li], xl, Mtiles, HID);
        k_gemm_wmma<<<gGemm, 256, 0, stream>>>(hh, Wrt[li], br[li], xr, Mtiles, HID);
        k_init_seg<<<nElem / 256, 256, 0, stream>>>(acc, emax, denom, nElem, NN * NHEAD);
        k_attn_score<<<edgeBlk, 256, 0, stream>>>(xl, xr, att[li], ei, ebuf, emax);
        k_attn_scatter<<<edgeBlk, 256, 0, stream>>>(xl, ebuf, emax, ei, denom, acc);
        k_node_finish<<<nElem / 256, 256, 0, stream>>>(acc, denom, bo[li], hcur, (li < 2) ? 1 : 0);
    }

    // final node features -> f16, then fused per-edge MLP (400000/16 = 25000 tiles exactly)
    k_f32_to_f16<<<nElem / 256, 256, 0, stream>>>(hcur, hh, nElem);
    k_edge_mlp<<<EE / 16, 256, 0, stream>>>(hh, ei, W1t, be1, W2t, be2, We3, be3, out);
}
